// CausalAttentionLayer_90666759619179
// MI455X (gfx1250) — matrix operand order repaired and checked by reading, not compiled, not run on hardware
//
#include <hip/hip_runtime.h>

typedef _Float16 half8  __attribute__((ext_vector_type(8)));
typedef _Float16 half16 __attribute__((ext_vector_type(16)));
typedef float    v8f    __attribute__((ext_vector_type(8)));

constexpr int Bb = 4, Ss = 2048, Ee = 1024, Hh = 16, Dd = 64;

__device__ __forceinline__ half16 cat8(half8 lo, half8 hi) {
  return __builtin_shufflevector(lo, hi, 0, 1, 2, 3, 4, 5, 6, 7,
                                 8, 9, 10, 11, 12, 13, 14, 15);
}

__device__ __forceinline__ v8f wmma_f16(half16 a, half16 b, v8f c) {
  // D = A(16x32 f16) * B(32x16 f16) + C(16x16 f32)
  return __builtin_amdgcn_wmma_f32_16x16x32_f16(false, a, false, b,
                                                (short)0, c, false, false);
}

// ---------------------------------------------------------------------------
// Weight transpose + fp32->fp16: W[K,N] -> Wt[N,K]
// ---------------------------------------------------------------------------
__global__ __launch_bounds__(256) void transpose_w(const float* __restrict__ W,
                                                   _Float16* __restrict__ Wt,
                                                   int K, int N) {
  size_t idx = (size_t)blockIdx.x * 256 + threadIdx.x;  // over N*K
  int k = (int)(idx % K);
  int n = (int)(idx / K);
  Wt[idx] = (_Float16)W[(size_t)k * N + n];
}

// ---------------------------------------------------------------------------
// Repack V from qkv[B*S, 3E] (V at col 2E + h*D + d) into Vt[B*H, D, S]
// ---------------------------------------------------------------------------
__global__ __launch_bounds__(256) void repack_vt(const _Float16* __restrict__ qkv,
                                                 _Float16* __restrict__ Vt) {
  size_t idx = (size_t)blockIdx.x * 256 + threadIdx.x;  // over B*H*D*S
  int s  = (int)(idx % Ss);
  int d  = (int)((idx / Ss) % Dd);
  int bh = (int)(idx / ((size_t)Ss * Dd));
  int b = bh / Hh, h = bh - b * Hh;
  Vt[idx] = qkv[(size_t)(b * Ss + s) * (3 * Ee) + 2 * Ee + h * Dd + d];
}

// ---------------------------------------------------------------------------
// WMMA GEMM: C[M,N] = A[M,K] * Bt[N,K]^T + bias[N]
// Block tile 128x128, K-step 32. 8 waves, each wave 64x32 (4x2 WMMA frags).
// Ping-pong LDS (2x16KB) with register-staged global prefetch: the tile for
// step i+1 is in flight while the 8 WMMAs of step i execute; one barrier per
// K-step. Tile i+2 is touched with global_prefetch_b8.
// ---------------------------------------------------------------------------
template <typename AT, typename OT>
__global__ __launch_bounds__(256) void gemm128(const AT* __restrict__ A,
                                               const _Float16* __restrict__ Bt,
                                               const float* __restrict__ bias,
                                               OT* __restrict__ C,
                                               int M, int N, int K) {
  __shared__ __align__(64) _Float16 As[2][128 * 32];
  __shared__ __align__(64) _Float16 Bs[2][128 * 32];
  const int tid   = threadIdx.x;
  const int lane  = tid & 31;
  const int wave  = tid >> 5;
  const int lrow  = lane & 15;
  const int khalf = (lane >> 4) << 3;  // unused; A-frag also uses koff
  const int koff  = (lane >> 4) << 4;  // B-frag K chunk base (contiguous 16)
  const int rofs  = (lane >> 4) << 3;  // C/D row offset
  const int m0 = blockIdx.y * 128;
  const int n0 = blockIdx.x * 128;
  const int wm = (wave >> 2) << 6;  // 0 / 64
  const int wn = (wave & 3) << 5;   // 0..96

  // Staging chunk coordinates: 512 chunks of 8 halves, 2 per thread.
  int rr[2], kk[2];
  rr[0] = tid >> 2;
  kk[0] = (tid & 3) << 3;
  rr[1] = (tid + 256) >> 2;
  kk[1] = ((tid + 256) & 3) << 3;

  v8f   ar[2];   // staged A chunks (fp32 source)
  half8 arh[2];  // staged A chunks (fp16 source)
  half8 br[2];   // staged B chunks

  auto gload = [&](int k0) {
#pragma unroll
    for (int c = 0; c < 2; ++c) {
      if constexpr (sizeof(AT) == 4) {
        ar[c] = *(const v8f*)((const float*)A + (size_t)(m0 + rr[c]) * K +
                              (k0 + kk[c]));
      } else {
        arh[c] = *(const half8*)((const _Float16*)A +
                                 (size_t)(m0 + rr[c]) * K + (k0 + kk[c]));
      }
      br[c] = *(const half8*)(Bt + (size_t)(n0 + rr[c]) * K + (k0 + kk[c]));
    }
  };
  auto sstore = [&](int buf) {
#pragma unroll
    for (int c = 0; c < 2; ++c) {
      _Float16* da = &As[buf][rr[c] * 32 + kk[c]];
      if constexpr (sizeof(AT) == 4) {
#pragma unroll
        for (int i = 0; i < 8; ++i) da[i] = (_Float16)ar[c][i];
      } else {
        *(half8*)da = arh[c];
      }
      *(half8*)&Bs[buf][rr[c] * 32 + kk[c]] = br[c];
    }
  };

  v8f acc[4][2] = {};
  const int nk = K >> 5;

  gload(0);
  sstore(0);
  __syncthreads();

  int cur = 0;
  for (int i = 0; i < nk; ++i) {
    if (i + 1 < nk) gload((i + 1) << 5);  // overlap with WMMAs below
    if (i + 2 < nk) {
      const int kp = (i + 2) << 5;
      __builtin_prefetch((const char*)A +
                             ((size_t)(m0 + rr[0]) * K + kp) * sizeof(AT),
                         0, 3);
      __builtin_prefetch(Bt + (size_t)(n0 + rr[0]) * K + kp, 0, 3);
    }

    half16 af[4], bf[2];
#pragma unroll
    for (int ii = 0; ii < 4; ++ii) {
      const _Float16* p = &As[cur][(wm + ii * 16 + lrow) * 32];
      af[ii] =
          cat8(*(const half8*)(p + koff), *(const half8*)(p + koff + 8));
    }
#pragma unroll
    for (int j = 0; j < 2; ++j) {
      const _Float16* p = &Bs[cur][(wn + j * 16 + lrow) * 32];
      bf[j] = *(const half16*)(p + koff);
    }
#pragma unroll
    for (int ii = 0; ii < 4; ++ii)
#pragma unroll
      for (int j = 0; j < 2; ++j)
        acc[ii][j] = wmma_f16(af[ii], bf[j], acc[ii][j]);

    if (i + 1 < nk) {
      sstore(cur ^ 1);   // disjoint buffer from the one just read
      __syncthreads();   // publish for next step / retire this step's reads
      cur ^= 1;
    }
  }

  // Epilogue: C-layout row = tile + r + 8*(lane>=16), col = tile + lane%16.
#pragma unroll
  for (int j = 0; j < 2; ++j) {
    int col  = n0 + wn + j * 16 + lrow;
    float bv = bias[col];
#pragma unroll
    for (int i = 0; i < 4; ++i) {
      int rowb = m0 + wm + i * 16 + rofs;
#pragma unroll
      for (int r = 0; r < 8; ++r) {
        float v = acc[i][j][r] + bv;
        C[(size_t)(rowb + r) * N + col] = (OT)v;
      }
    }
  }
}

// ---------------------------------------------------------------------------
// Flash attention: one wave handles a 16-query block of one (b,h).
// Online softmax over 32-key blocks; S and O via WMMA; P repacked via LDS.
// ---------------------------------------------------------------------------
__global__ __launch_bounds__(128) void attn_fa(const _Float16* __restrict__ qkv,
                                               const _Float16* __restrict__ Vt,
                                               _Float16* __restrict__ O) {
  __shared__ __align__(64) _Float16 Plds[4][16 * 32];
  const int lane  = threadIdx.x & 31;
  const int wv    = threadIdx.x >> 5;
  const int lrow  = lane & 15;
  const int khalf = (lane >> 4) << 3;
  const int koff  = (lane >> 4) << 4;
  const int rofs  = (lane >> 4) << 3;

  const int gw = blockIdx.x * 4 + wv;
  const int QB = Ss / 16;
  const int bh = gw / QB;
  const int qb = gw - bh * QB;
  const int b  = bh / Hh, h = bh - b * Hh;
  const int q0 = qb * 16;

  const _Float16* vhead = Vt + (size_t)bh * Dd * Ss;

  // Q A-fragments (d = 0..31 and 32..63)
  const _Float16* qrow =
      qkv + (size_t)(b * Ss + q0 + lrow) * (3 * Ee) + h * Dd;
  half16 qf0 = cat8(*(const half8*)(qrow + koff),
                    *(const half8*)(qrow + koff + 8));
  half16 qf1 = cat8(*(const half8*)(qrow + 32 + koff),
                    *(const half8*)(qrow + 32 + koff + 8));

  float m[8], l[8];
  v8f o[4] = {};
#pragma unroll
  for (int r = 0; r < 8; ++r) { m[r] = -3.0e38f; l[r] = 0.0f; }

  const int nkb = (q0 >> 5) + 1;  // key blocks covering keys 0..q0+15
  for (int kb = 0; kb < nkb; ++kb) {
    const int k0 = kb << 5;

    if (kb + 1 < nkb) {  // pull next K/V block toward the caches
      const int kn = k0 + 32;
      __builtin_prefetch(
          qkv + (size_t)(b * Ss + kn + lane) * (3 * Ee) + Ee + h * Dd, 0, 3);
      __builtin_prefetch(vhead + (size_t)lane * Ss + kn, 0, 3);
      __builtin_prefetch(vhead + (size_t)(32 + lane) * Ss + kn, 0, 3);
    }

    v8f sc[2] = {};
#pragma unroll
    for (int t = 0; t < 2; ++t) {
      const _Float16* krow =
          qkv + (size_t)(b * Ss + k0 + t * 16 + lrow) * (3 * Ee) + Ee + h * Dd;
      half16 kb0 = *(const half16*)(krow + koff);
      half16 kb1 = *(const half16*)(krow + 32 + koff);
      sc[t] = wmma_f16(qf0, kb0, sc[t]);
      sc[t] = wmma_f16(qf1, kb1, sc[t]);
    }

    const float scale = 0.125f;            // 1/sqrt(64)
    const bool needmask = (k0 + 31 > q0);  // wave-uniform predicate
    float mn[8], rs[8];
#pragma unroll
    for (int r = 0; r < 8; ++r) {
      float v0 = sc[0][r] * scale;
      float v1 = sc[1][r] * scale;
      if (needmask) {
        int row = q0 + r + rofs;
        if (k0 + lrow > row)      v0 = -3.0e38f;
        if (k0 + 16 + lrow > row) v1 = -3.0e38f;
      }
      sc[0][r] = v0;
      sc[1][r] = v1;
      float v = fmaxf(v0, v1);
      v = fmaxf(v, __shfl_xor(v, 1, 32));
      v = fmaxf(v, __shfl_xor(v, 2, 32));
      v = fmaxf(v, __shfl_xor(v, 4, 32));
      v = fmaxf(v, __shfl_xor(v, 8, 32));
      mn[r] = fmaxf(m[r], v);
    }
#pragma unroll
    for (int r = 0; r < 8; ++r) {
      float p0 = __expf(sc[0][r] - mn[r]);
      float p1 = __expf(sc[1][r] - mn[r]);
      sc[0][r] = p0;
      sc[1][r] = p1;
      float v = p0 + p1;
      v += __shfl_xor(v, 1, 32);
      v += __shfl_xor(v, 2, 32);
      v += __shfl_xor(v, 4, 32);
      v += __shfl_xor(v, 8, 32);
      rs[r] = v;
    }
#pragma unroll
    for (int r = 0; r < 8; ++r) {
      float cf = __expf(m[r] - mn[r]);
      l[r] = l[r] * cf + rs[r];
      m[r] = mn[r];
      o[0][r] *= cf; o[1][r] *= cf; o[2][r] *= cf; o[3][r] *= cf;
    }

    // Repack P (C-layout f32) -> 16x32 fp16 tile in wave-private LDS.
    _Float16* pl = Plds[wv];
#pragma unroll
    for (int r = 0; r < 8; ++r) {
      pl[(r + rofs) * 32 + lrow]      = (_Float16)sc[0][r];
      pl[(r + rofs) * 32 + 16 + lrow] = (_Float16)sc[1][r];
    }
    asm volatile("s_wait_dscnt 0" ::: "memory");  // wave-local store->load order
    const _Float16* pr = pl + lrow * 32;
    half16 pa = cat8(*(const half8*)(pr + koff),
                     *(const half8*)(pr + koff + 8));

    // O += P * V  (V^T rows contiguous in key dim via Vt[B*H, D, S])
#pragma unroll
    for (int n = 0; n < 4; ++n) {
      const _Float16* vrow = vhead + (size_t)(n * 16 + lrow) * Ss + k0;
      half16 vb = *(const half16*)(vrow + koff);
      o[n] = wmma_f16(pa, vb, o[n]);
    }
  }

  // Finalize: O / l, write fp16 head-interleaved -> [B*S, E]
#pragma unroll
  for (int r = 0; r < 8; ++r) {
    float inv = 1.0f / l[r];
    int row = q0 + r + rofs;
    _Float16* op = O + (size_t)(b * Ss + row) * Ee + h * Dd + lrow;
    op[0]  = (_Float16)(o[0][r] * inv);
    op[16] = (_Float16)(o[1][r] * inv);
    op[32] = (_Float16)(o[2][r] * inv);
    op[48] = (_Float16)(o[3][r] * inv);
  }
}

// ---------------------------------------------------------------------------
extern "C" void kernel_launch(void* const* d_in, const int* in_sizes, int n_in,
                              void* d_out, int out_size, void* d_ws,
                              size_t ws_size, hipStream_t stream) {
  const float* x     = (const float*)d_in[0];
  const float* W_qkv = (const float*)d_in[1];
  const float* b_qkv = (const float*)d_in[2];
  const float* W_out = (const float*)d_in[3];
  const float* b_out = (const float*)d_in[4];
  float* out = (float*)d_out;

  char* ws = (char*)d_ws;
  size_t off = 0;
  _Float16* Wqkv_t = (_Float16*)(ws + off); off += (size_t)3 * Ee * Ee * 2;
  _Float16* Wout_t = (_Float16*)(ws + off); off += (size_t)Ee * Ee * 2;
  _Float16* qkvb   = (_Float16*)(ws + off); off += (size_t)Bb * Ss * 3 * Ee * 2;
  _Float16* Vt     = (_Float16*)(ws + off); off += (size_t)Bb * Hh * Dd * Ss * 2;
  _Float16* Obuf   = (_Float16*)(ws + off); off += (size_t)Bb * Ss * Ee * 2;
  (void)ws_size; (void)in_sizes; (void)n_in; (void)out_size;

  transpose_w<<<(3 * Ee * Ee) / 256, 256, 0, stream>>>(W_qkv, Wqkv_t, Ee, 3 * Ee);
  transpose_w<<<(Ee * Ee) / 256, 256, 0, stream>>>(W_out, Wout_t, Ee, Ee);

  gemm128<float, _Float16>
      <<<dim3((3 * Ee) / 128, (Bb * Ss) / 128), 256, 0, stream>>>(
          x, Wqkv_t, b_qkv, qkvb, Bb * Ss, 3 * Ee, Ee);

  repack_vt<<<(Bb * Hh * Dd * Ss) / 256, 256, 0, stream>>>(qkvb, Vt);

  attn_fa<<<(Bb * Hh * (Ss / 16)) / 4, 128, 0, stream>>>(qkvb, Vt, Obuf);

  gemm128<_Float16, float>
      <<<dim3(Ee / 128, (Bb * Ss) / 128), 256, 0, stream>>>(
          Obuf, Wout_t, b_out, out, Bb * Ss, Ee, Ee);
}
